// Qwen2MemDecoderLayer_13374528160290
// MI455X (gfx1250) — matrix operand order repaired and checked by reading, not compiled, not run on hardware
//
#include <hip/hip_runtime.h>
#include <hip/hip_bf16.h>

typedef __attribute__((ext_vector_type(16))) __bf16 bf16x16;
typedef __attribute__((ext_vector_type(8)))  float  f32x8;
typedef unsigned short u16;
typedef unsigned int   u32;
typedef __attribute__((ext_vector_type(8)))  u16 u16x8;
typedef __attribute__((ext_vector_type(16))) u16 u16x16;

#define DEV __device__ __forceinline__

#define BATCH 2
#define SEQ   2048
#define DMODEL 1024
#define NH    16
#define NKV   4
#define HDIM  64
#define FFDIM 4096
#define WIN   1024

// Native bf16 convert (RTNE) — lets the compiler use hardware cvt ops.
DEV u16 f2bf(float f) {
  __bf16 h = (__bf16)f;
  return __builtin_bit_cast(u16, h);
}
DEV u32 f2bf_pk(float lo, float hi) {
  return (u32)f2bf(lo) | ((u32)f2bf(hi) << 16);
}
DEV float fastrcp(float x) { return __builtin_amdgcn_rcpf(x); }

DEV bf16x16 pack_frag(u16x8 lo, u16x8 hi) {
  u16x16 t;
#pragma unroll
  for (int e = 0; e < 8; e++) { t[e] = lo[e]; t[8 + e] = hi[e]; }
  return __builtin_bit_cast(bf16x16, t);
}

DEV f32x8 wmma_bf16(bf16x16 a, bf16x16 b, f32x8 c) {
  return __builtin_amdgcn_wmma_f32_16x16x32_bf16(false, a, false, b, (short)0, c,
                                                 false, false);
}

// ---------------------------------------------------------------------------
// RMSNorm: one block per row (D=1024), writes bf16 (u16) matrix for GEMM A.
// ---------------------------------------------------------------------------
__global__ __launch_bounds__(256) void rmsnorm_kernel(
    const float* __restrict__ x, const float* __restrict__ w,
    u16* __restrict__ out) {
  __shared__ float red[8];
  const int row = blockIdx.x;
  const int tid = threadIdx.x, lane = tid & 31, wid = tid >> 5;
  const float* xr = x + (size_t)row * DMODEL;
  float4 xv = *(const float4*)(xr + tid * 4);
  float ss = xv.x * xv.x + xv.y * xv.y + xv.z * xv.z + xv.w * xv.w;
#pragma unroll
  for (int m = 1; m < 32; m <<= 1) ss += __shfl_xor(ss, m, 32);
  if (lane == 0) red[wid] = ss;
  __syncthreads();
  if (wid == 0) {
    float t = red[lane & 7];
#pragma unroll
    for (int m = 1; m < 8; m <<= 1) t += __shfl_xor(t, m, 32);
    if (lane == 0) red[0] = t;
  }
  __syncthreads();
  const float scale = rsqrtf(red[0] * (1.0f / (float)DMODEL) + 1e-6f);
  float4 wv = *(const float4*)(w + tid * 4);
  uint2 pk;
  pk.x = f2bf_pk(xv.x * scale * wv.x, xv.y * scale * wv.y);
  pk.y = f2bf_pk(xv.z * scale * wv.z, xv.w * scale * wv.w);
  *(uint2*)(out + (size_t)row * DMODEL + tid * 4) = pk;
}

// ---------------------------------------------------------------------------
// RoPE, in place on q or k buffer laid out [B,S,nh,64]. One thread per (d,d+32).
// ---------------------------------------------------------------------------
__global__ void rope_kernel(float* __restrict__ x, const float* __restrict__ cosb,
                            const float* __restrict__ sinb, int nheads) {
  int idx = blockIdx.x * blockDim.x + threadIdx.x;
  int d = idx & 31;
  int t = idx >> 5;
  int hh = t % nheads; t /= nheads;
  int s = t % SEQ;     int b = t / SEQ;
  size_t base = ((size_t)(b * SEQ + s) * nheads + hh) * HDIM;
  float x1 = x[base + d], x2 = x[base + d + 32];
  float c = cosb[s * HDIM + d], sn = sinb[s * HDIM + d];
  x[base + d]      = x1 * c - x2 * sn;
  x[base + d + 32] = x2 * c + x1 * sn;
}

// ---------------------------------------------------------------------------
// Generic tiled GEMM: C[M,N] = A(bf16)[M,K] * B(f32)[K,N] (+bias)(+resid) -> f32
// Block tile 64x128, 8 waves, wave tile 32x32 (2x2 wmma), K-step 32.
// HB/HR compile-time: straight-line epilogue, no per-element branches.
// ---------------------------------------------------------------------------
#define BM 64
#define BN 128
#define BK 32
#define LDT 40  // padded LDS row stride in u16 (80B: 16B aligned, conflict-free)

template <bool HB, bool HR>
__global__ __launch_bounds__(256) void gemm_bf16_epi(
    const u16* __restrict__ A, const float* __restrict__ Bw,
    const float* __restrict__ bias, const float* __restrict__ resid,
    float* __restrict__ C, int M, int N, int K) {
  __shared__ u16 As[BM * LDT];
  __shared__ u16 Bs[BN * LDT];
  const int tid = threadIdx.x, lane = tid & 31, wid = tid >> 5;
  const int m0 = blockIdx.y * BM, n0 = blockIdx.x * BN;
  const int wm = (wid >> 2) * 32, wn = (wid & 3) * 32;
  const int nl = lane & 15, g = lane >> 4;

  f32x8 acc[2][2] = {};

  for (int kk = 0; kk < K; kk += BK) {
    {  // stage A tile (bf16): 64x32, 8 u16 per thread
      int r = tid >> 2, c = (tid & 3) * 8;
      u16x8 va = *(const u16x8*)(A + (size_t)(m0 + r) * K + kk + c);
      *(u16x8*)(&As[r * LDT + c]) = va;
    }
#pragma unroll
    for (int i = 0; i < 2; i++) {  // stage B f32 -> bf16, transposed, packed b32
      int idx = i * 256 + tid;     // 512 items: 2 k-rows x 4 n each
      int r0 = (idx >> 5) * 2, c4 = (idx & 31) * 4;
      float4 v0 = *(const float4*)(Bw + (size_t)(kk + r0) * N + n0 + c4);
      float4 v1 = *(const float4*)(Bw + (size_t)(kk + r0 + 1) * N + n0 + c4);
      *(u32*)(&Bs[(c4 + 0) * LDT + r0]) = f2bf_pk(v0.x, v1.x);
      *(u32*)(&Bs[(c4 + 1) * LDT + r0]) = f2bf_pk(v0.y, v1.y);
      *(u32*)(&Bs[(c4 + 2) * LDT + r0]) = f2bf_pk(v0.z, v1.z);
      *(u32*)(&Bs[(c4 + 3) * LDT + r0]) = f2bf_pk(v0.w, v1.w);
    }
    __syncthreads();
    bf16x16 af[2], bfg[2];
#pragma unroll
    for (int a = 0; a < 2; a++) {  // A fragment: lane-half K pattern
      const u16* p = &As[(wm + a * 16 + nl) * LDT + g * 16];
      af[a] = pack_frag(*(const u16x8*)p, *(const u16x8*)(p + 8));
    }
#pragma unroll
    for (int b = 0; b < 2; b++) {  // B fragment: K = halflane*16 + e
      const u16* p = &Bs[(wn + b * 16 + nl) * LDT + g * 16];
      bfg[b] = pack_frag(*(const u16x8*)p, *(const u16x8*)(p + 8));
    }
#pragma unroll
    for (int a = 0; a < 2; a++)
#pragma unroll
      for (int b = 0; b < 2; b++) acc[a][b] = wmma_bf16(af[a], bfg[b], acc[a][b]);
    __syncthreads();
  }

#pragma unroll
  for (int a = 0; a < 2; a++)
#pragma unroll
    for (int b = 0; b < 2; b++) {
      const int gc = n0 + wn + b * 16 + nl;
      const float bv = HB ? bias[gc] : 0.0f;
      const int rbase = m0 + wm + a * 16 + 8 * g;
#pragma unroll
      for (int v = 0; v < 8; v++) {
        const size_t off = (size_t)(rbase + v) * N + gc;
        float val = acc[a][b][v] + bv;
        if (HR) val += resid[off];
        C[off] = val;
      }
    }
}

// ---------------------------------------------------------------------------
// Fused gate/up GEMM: out(bf16) = silu(A*Wg) * (A*Wu)
// ---------------------------------------------------------------------------
__global__ __launch_bounds__(256) void gemm_gateup(
    const u16* __restrict__ A, const float* __restrict__ Bg,
    const float* __restrict__ Bu, u16* __restrict__ Out, int M, int N, int K) {
  __shared__ u16 As[BM * LDT];
  __shared__ u16 Bsg[BN * LDT];
  __shared__ u16 Bsu[BN * LDT];
  const int tid = threadIdx.x, lane = tid & 31, wid = tid >> 5;
  const int m0 = blockIdx.y * BM, n0 = blockIdx.x * BN;
  const int wm = (wid >> 2) * 32, wn = (wid & 3) * 32;
  const int nl = lane & 15, g = lane >> 4;

  f32x8 accg[2][2] = {};
  f32x8 accu[2][2] = {};

  for (int kk = 0; kk < K; kk += BK) {
    {
      int r = tid >> 2, c = (tid & 3) * 8;
      u16x8 va = *(const u16x8*)(A + (size_t)(m0 + r) * K + kk + c);
      *(u16x8*)(&As[r * LDT + c]) = va;
    }
#pragma unroll
    for (int i = 0; i < 2; i++) {
      int idx = i * 256 + tid;
      int r0 = (idx >> 5) * 2, c4 = (idx & 31) * 4;
      float4 g0 = *(const float4*)(Bg + (size_t)(kk + r0) * N + n0 + c4);
      float4 g1 = *(const float4*)(Bg + (size_t)(kk + r0 + 1) * N + n0 + c4);
      float4 u0 = *(const float4*)(Bu + (size_t)(kk + r0) * N + n0 + c4);
      float4 u1 = *(const float4*)(Bu + (size_t)(kk + r0 + 1) * N + n0 + c4);
      *(u32*)(&Bsg[(c4 + 0) * LDT + r0]) = f2bf_pk(g0.x, g1.x);
      *(u32*)(&Bsg[(c4 + 1) * LDT + r0]) = f2bf_pk(g0.y, g1.y);
      *(u32*)(&Bsg[(c4 + 2) * LDT + r0]) = f2bf_pk(g0.z, g1.z);
      *(u32*)(&Bsg[(c4 + 3) * LDT + r0]) = f2bf_pk(g0.w, g1.w);
      *(u32*)(&Bsu[(c4 + 0) * LDT + r0]) = f2bf_pk(u0.x, u1.x);
      *(u32*)(&Bsu[(c4 + 1) * LDT + r0]) = f2bf_pk(u0.y, u1.y);
      *(u32*)(&Bsu[(c4 + 2) * LDT + r0]) = f2bf_pk(u0.z, u1.z);
      *(u32*)(&Bsu[(c4 + 3) * LDT + r0]) = f2bf_pk(u0.w, u1.w);
    }
    __syncthreads();
    bf16x16 af[2], bg[2], bu[2];
#pragma unroll
    for (int a = 0; a < 2; a++) {
      const u16* p = &As[(wm + a * 16 + nl) * LDT + g * 16];
      af[a] = pack_frag(*(const u16x8*)p, *(const u16x8*)(p + 8));
    }
#pragma unroll
    for (int b = 0; b < 2; b++) {
      const u16* pg = &Bsg[(wn + b * 16 + nl) * LDT + g * 16];
      const u16* pu = &Bsu[(wn + b * 16 + nl) * LDT + g * 16];
      bg[b] = pack_frag(*(const u16x8*)pg, *(const u16x8*)(pg + 8));
      bu[b] = pack_frag(*(const u16x8*)pu, *(const u16x8*)(pu + 8));
    }
#pragma unroll
    for (int a = 0; a < 2; a++)
#pragma unroll
      for (int b = 0; b < 2; b++) {
        accg[a][b] = wmma_bf16(af[a], bg[b], accg[a][b]);
        accu[a][b] = wmma_bf16(af[a], bu[b], accu[a][b]);
      }
    __syncthreads();
  }

#pragma unroll
  for (int a = 0; a < 2; a++)
#pragma unroll
    for (int b = 0; b < 2; b++) {
      const int gc = n0 + wn + b * 16 + nl;
      const int rbase = m0 + wm + a * 16 + 8 * g;
#pragma unroll
      for (int v = 0; v < 8; v++) {
        float gv = accg[a][b][v];
        // silu via v_exp + v_rcp (avoid IEEE divide expansion)
        float sg = gv * fastrcp(1.0f + __expf(-gv));
        Out[(size_t)(rbase + v) * N + gc] = f2bf(sg * accu[a][b][v]);
      }
    }
}

// ---------------------------------------------------------------------------
// Flash attention (GQA, sliding window causal). One wave per 16-query tile.
// Key blocks of 32. Scores: Q(16x64)*K^T via 2x wmma per 16-key half.
// P routed through per-wave LDS tile to convert C-layout -> A-layout.
// ---------------------------------------------------------------------------
__global__ __launch_bounds__(128) void attn_kernel(
    const float* __restrict__ q, const float* __restrict__ k,
    const float* __restrict__ v, u16* __restrict__ ctxb) {
  __shared__ u16 plds[4][16 * 32];
  const int lane = threadIdx.x & 31, wid = threadIdx.x >> 5;
  const int qt = blockIdx.x * 4 + wid;
  const int h = blockIdx.y, b = blockIdx.z;
  const int q0 = qt * 16;
  const int hkv = h / (NH / NKV);
  const int nl = lane & 15, g = lane >> 4;

  // Q fragments with 1/sqrt(HD) folded into conversion
  bf16x16 qf[2];
  {
    const float scl = 0.125f;
    const float* qp = q + ((size_t)(b * SEQ + q0 + nl) * NH + h) * HDIM;
#pragma unroll
    for (int ck = 0; ck < 2; ck++) {
      const float* p = qp + ck * 32 + g * 16;
      u16x16 t;
#pragma unroll
      for (int e = 0; e < 8; e++) t[e] = f2bf(p[e] * scl);
#pragma unroll
      for (int e = 0; e < 8; e++) t[8 + e] = f2bf(p[8 + e] * scl);
      qf[ck] = __builtin_bit_cast(bf16x16, t);
    }
  }

  f32x8 ctx[4] = {};
  float mrow[8], lrow[8], corr[8], pcur[2][8];
#pragma unroll
  for (int vv = 0; vv < 8; vv++) { mrow[vv] = -INFINITY; lrow[vv] = 0.0f; }

  int jmin = q0 - WIN + 1;
  if (jmin < 0) jmin = 0;
  for (int j0 = jmin & ~31; j0 <= q0 + 15; j0 += 32) {
    // ---- scores: two 16x16 tiles over the 32-key block
    f32x8 sc[2] = {};
#pragma unroll
    for (int kh = 0; kh < 2; kh++) {
      const float* kp =
          k + ((size_t)(b * SEQ + j0 + kh * 16 + nl) * NKV + hkv) * HDIM;
#pragma unroll
      for (int ck = 0; ck < 2; ck++) {
        const float* p = kp + ck * 32 + g * 16;  // B frag: K(hd)=g*16+e
        u16x16 t;
#pragma unroll
        for (int e = 0; e < 16; e++) t[e] = f2bf(p[e]);
        sc[kh] = wmma_bf16(qf[ck], __builtin_bit_cast(bf16x16, t), sc[kh]);
      }
    }
    // ---- online softmax (row reductions across 16-lane halves)
#pragma unroll
    for (int vv = 0; vv < 8; vv++) {
      int irow = q0 + vv + 8 * g;
      int jc0 = j0 + nl, jc1 = j0 + 16 + nl;
      bool ok0 = (jc0 <= irow) && (irow - jc0 < WIN);
      bool ok1 = (jc1 <= irow) && (irow - jc1 < WIN);
      float s0 = ok0 ? sc[0][vv] : -INFINITY;
      float s1 = ok1 ? sc[1][vv] : -INFINITY;
      float t = fmaxf(s0, s1);
#pragma unroll
      for (int m = 1; m < 16; m <<= 1) t = fmaxf(t, __shfl_xor(t, m, 32));
      float mn = fmaxf(mrow[vv], t);
      float c = (mn == -INFINITY) ? 1.0f : __expf(mrow[vv] - mn);
      float p0 = ok0 ? __expf(s0 - mn) : 0.0f;
      float p1 = ok1 ? __expf(s1 - mn) : 0.0f;
      float rs = p0 + p1;
#pragma unroll
      for (int m = 1; m < 16; m <<= 1) rs += __shfl_xor(rs, m, 32);
      lrow[vv] = lrow[vv] * c + rs;
      mrow[vv] = mn;
      corr[vv] = c;
      pcur[0][vv] = p0;
      pcur[1][vv] = p1;
    }
#pragma unroll
    for (int c4 = 0; c4 < 4; c4++)
#pragma unroll
      for (int vv = 0; vv < 8; vv++) ctx[c4][vv] *= corr[vv];
    // ---- C-layout -> A-layout via per-wave LDS tile
    u16* pl = plds[wid];
#pragma unroll
    for (int vv = 0; vv < 8; vv++) {
      pl[(vv + 8 * g) * 32 + nl] = f2bf(pcur[0][vv]);
      pl[(vv + 8 * g) * 32 + 16 + nl] = f2bf(pcur[1][vv]);
    }
    asm volatile("s_wait_dscnt 0x0" ::: "memory");
    bf16x16 pf;
    {
      const u16* p = pl + nl * 32 + g * 16;
      pf = pack_frag(*(const u16x8*)p, *(const u16x8*)(p + 8));
    }
    // ---- ctx += P(16x32) * V(32x64), N chunks of 16
#pragma unroll
    for (int c4 = 0; c4 < 4; c4++) {
      const float* vp = v + ((size_t)(b * SEQ + j0 + g * 16) * NKV + hkv) * HDIM +
                        c4 * 16 + nl;
      u16x16 t;
#pragma unroll
      for (int e = 0; e < 16; e++) t[e] = f2bf(vp[(size_t)e * NKV * HDIM]);
      ctx[c4] = wmma_bf16(pf, __builtin_bit_cast(bf16x16, t), ctx[c4]);
    }
  }
  // ---- normalize (rcp, not divide) + store bf16 context [B,S,H*HD]
  float rinv[8];
#pragma unroll
  for (int vv = 0; vv < 8; vv++) rinv[vv] = fastrcp(lrow[vv]);
#pragma unroll
  for (int c4 = 0; c4 < 4; c4++)
#pragma unroll
    for (int vv = 0; vv < 8; vv++) {
      int row = q0 + vv + 8 * g;
      ctxb[(size_t)(b * SEQ + row) * (NH * HDIM) + h * HDIM + c4 * 16 + nl] =
          f2bf(ctx[c4][vv] * rinv[vv]);
    }
}

// ---------------------------------------------------------------------------
extern "C" void kernel_launch(void* const* d_in, const int* in_sizes, int n_in,
                              void* d_out, int out_size, void* d_ws,
                              size_t ws_size, hipStream_t stream) {
  const float* hs   = (const float*)d_in[0];
  const float* cosb = (const float*)d_in[1];
  const float* sinb = (const float*)d_in[2];
  const float* wq   = (const float*)d_in[3];
  const float* bq   = (const float*)d_in[4];
  const float* wk   = (const float*)d_in[5];
  const float* bk   = (const float*)d_in[6];
  const float* wv   = (const float*)d_in[7];
  const float* bv   = (const float*)d_in[8];
  const float* wo   = (const float*)d_in[9];
  const float* ln1  = (const float*)d_in[10];
  const float* ln2  = (const float*)d_in[11];
  const float* wg   = (const float*)d_in[12];
  const float* wu   = (const float*)d_in[13];
  const float* wd   = (const float*)d_in[14];
  float* out = (float*)d_out;

  const int M = BATCH * SEQ;  // 4096
  char* ws = (char*)d_ws;
  u16* hn1    = (u16*)ws;   ws += (size_t)M * DMODEL * 2;
  float* qb   = (float*)ws; ws += (size_t)M * NH * HDIM * 4;
  float* kb   = (float*)ws; ws += (size_t)M * NKV * HDIM * 4;
  float* vb   = (float*)ws; ws += (size_t)M * NKV * HDIM * 4;
  u16* ctxb   = (u16*)ws;   ws += (size_t)M * NH * HDIM * 2;
  float* h1   = (float*)ws; ws += (size_t)M * DMODEL * 4;
  u16* hn2    = (u16*)ws;   ws += (size_t)M * DMODEL * 2;
  u16* act    = (u16*)ws;   ws += (size_t)M * FFDIM * 2;

  dim3 blk(256);

  // 1) RMSNorm -> bf16
  rmsnorm_kernel<<<M, 256, 0, stream>>>(hs, ln1, hn1);
  // 2) QKV projections (+bias)
  gemm_bf16_epi<true, false><<<dim3((NH * HDIM) / BN, M / BM), blk, 0, stream>>>(
      hn1, wq, bq, nullptr, qb, M, NH * HDIM, DMODEL);
  gemm_bf16_epi<true, false><<<dim3((NKV * HDIM) / BN, M / BM), blk, 0, stream>>>(
      hn1, wk, bk, nullptr, kb, M, NKV * HDIM, DMODEL);
  gemm_bf16_epi<true, false><<<dim3((NKV * HDIM) / BN, M / BM), blk, 0, stream>>>(
      hn1, wv, bv, nullptr, vb, M, NKV * HDIM, DMODEL);
  // 3) RoPE on q and k
  rope_kernel<<<(BATCH * SEQ * NH * 32) / 256, 256, 0, stream>>>(qb, cosb, sinb, NH);
  rope_kernel<<<(BATCH * SEQ * NKV * 32) / 256, 256, 0, stream>>>(kb, cosb, sinb, NKV);
  // 4) Flash attention -> bf16 context
  attn_kernel<<<dim3(SEQ / 64, NH, BATCH), 128, 0, stream>>>(qb, kb, vb, ctxb);
  // 5) O projection + residual
  gemm_bf16_epi<false, true><<<dim3(DMODEL / BN, M / BM), blk, 0, stream>>>(
      ctxb, wo, nullptr, hs, h1, M, DMODEL, NH * HDIM);
  // 6) RMSNorm2 -> bf16
  rmsnorm_kernel<<<M, 256, 0, stream>>>(h1, ln2, hn2);
  // 7) fused gate/up + SwiGLU -> bf16
  gemm_gateup<<<dim3(FFDIM / BN, M / BM), blk, 0, stream>>>(hn2, wg, wu, act, M,
                                                            FFDIM, DMODEL);
  // 8) down projection + residual -> output
  gemm_bf16_epi<false, true><<<dim3(DMODEL / BN, M / BM), blk, 0, stream>>>(
      act, wd, nullptr, h1, out, M, DMODEL, FFDIM);
}
